// MODEL_FLUID_32246614459244
// MI455X (gfx1250) — matrix operand order repaired and checked by reading, not compiled, not run on hardware
//
#include <hip/hip_runtime.h>

typedef __attribute__((ext_vector_type(16))) _Float16 v16h;
typedef __attribute__((ext_vector_type(8)))  _Float16 v8h;
typedef __attribute__((ext_vector_type(8)))  float    v8f;
typedef __attribute__((ext_vector_type(4)))  unsigned int u32x4;
typedef __attribute__((ext_vector_type(8)))  int i32x8;
typedef __attribute__((ext_vector_type(4)))  int i32x4;

#define HD 256
#define NPTS 131072
#define TILE_P 16
#define NCH 6
#define THREADS 192

// Convert + transpose the four 256x256 f32 weight matrices into f16 [n][k]
// layout in d_ws so WMMA B-fragments are two runs of 8 contiguous halves per lane.
__global__ void cvt_weights(const float* __restrict__ Wu2, const float* __restrict__ Wu3,
                            const float* __restrict__ Wp2, const float* __restrict__ Wp3,
                            _Float16* __restrict__ out) {
    int idx = blockIdx.x * blockDim.x + threadIdx.x;
    if (idx >= 4 * HD * HD) return;
    int w = idx >> 16;            // which matrix
    int e = idx & (HD * HD - 1);  // element within matrix (k*256 + n)
    int k = e >> 8;
    int n = e & 255;
    const float* src = (w == 0) ? Wu2 : (w == 1) ? Wu3 : (w == 2) ? Wp2 : Wp3;
    out[w * HD * HD + n * HD + k] = (_Float16)src[e];
}

// ---- Tensor Data Mover: one-shot 256x256 x 2B tile, global -> LDS ----
// D# built per CDNA5 ISA (2-D tensor, groups 2/3 unused).
// amdgpu-toolchain (clang-23) 6-arg builtin form.
__device__ __forceinline__ void tdm_stage_W(const _Float16* src, _Float16* dstLDS) {
#if __has_builtin(__builtin_amdgcn_tensor_load_to_lds)
    const unsigned long long ga = (unsigned long long)(uintptr_t)src;
    const unsigned lds = (unsigned)(uintptr_t)dstLDS;  // low 32 bits = LDS byte offset
    u32x4 g0;
    g0.x = 1u;                                          // count=1, user-mode descriptor
    g0.y = lds;                                         // lds_addr [63:32]
    g0.z = (unsigned)ga;                                // global_addr [95:64]
    g0.w = ((unsigned)(ga >> 32) & 0x1FFFFFFu)          // global_addr [120:96]
         | (2u << 30);                                  // type=2 ("image")
    i32x8 g1;
    g1[0] = 0x00010000;                                 // data_size=1 (2 bytes)
    g1[1] = (int)(256u << 16);                          // tensor_dim0 = 256
    g1[2] = (int)(256u << 16);                          // tensor_dim1 = 256
    g1[3] = (int)(256u << 16);                          // tile_dim0   = 256
    g1[4] = 256;                                        // tile_dim1   = 256
    g1[5] = 256;                                        // tensor_dim0_stride = 256
    g1[6] = 0;
    g1[7] = 0;
    const i32x4 z4 = {0, 0, 0, 0};                      // groups 2/3: unused (2-D)
    const i32x8 z8 = {0, 0, 0, 0, 0, 0, 0, 0};          // extra group (unused)
    __builtin_amdgcn_tensor_load_to_lds(g0, g1, z4, z4, z8, 0);
    __builtin_amdgcn_s_wait_tensorcnt(0);
#else
    // fallback: manual cooperative copy (all lanes of calling wave)
    const int lane = (int)(threadIdx.x & 31);
    const uint4* s = (const uint4*)src;
    uint4*       d = (uint4*)dstLDS;
    for (int i = lane; i < (HD * HD * 2) / 16; i += 32) d[i] = s[i];
#endif
}

__global__ __launch_bounds__(THREADS, 1)
void pinn_residual_kernel(const float* __restrict__ X,
                          const float* __restrict__ Wu1, const float* __restrict__ bu1,
                          const float* __restrict__ bu2, const float* __restrict__ bu3,
                          const float* __restrict__ Wu4, const float* __restrict__ bu4,
                          const float* __restrict__ Wp1, const float* __restrict__ bp1,
                          const float* __restrict__ bp2, const float* __restrict__ bp3,
                          const float* __restrict__ Wp4,
                          const _Float16* __restrict__ WT,  // 4 transposed f16 matrices
                          float* __restrict__ out)
{
    // LDS: activations (f16), pre-activations (f32), staged weights (f16), results
    __shared__ __attribute__((aligned(32))) _Float16 sA[NCH * TILE_P * HD]; // 48 KB
    __shared__ __attribute__((aligned(16))) float    sZ[NCH * TILE_P * HD]; // 96 KB
    __shared__ __attribute__((aligned(32))) _Float16 sW[HD * HD];           // 128 KB
    __shared__ float sres[TILE_P][14];

    const int tid  = threadIdx.x;
    const int lane = tid & 31;
    // wave id, forced wave-uniform so all wv/ch branches become SALU branches
    const int wv   = __builtin_amdgcn_readfirstlane(tid >> 5);   // 0..5
    const int m    = lane & 15;      // row (A) / col-N (B,C,D) index
    const int hi   = lane >> 4;      // lane-half selector
    const int g0p  = blockIdx.x * TILE_P;

    // ---- 16x256 @ 256x256 GEMM for channel `ch`, N-tiles [ntLo,ntHi) ----
    // Two independent accumulator chains (K 0..127 / 128..255) to break the
    // WMMA->WMMA RAW dependency; summed (plus bias) at tile end.
    auto gemm = [&](int ch, int ntLo, int ntHi, const float* bias, bool use_bias) {
        const _Float16* arow = &sA[(ch * TILE_P + m) * HD];
        for (int nt = ntLo; nt < ntHi; ++nt) {
            const int n0 = nt * 16;
            const _Float16* bcol = &sW[(n0 + m) * HD];
            v8f acc0 = {}, acc1 = {};
            #pragma unroll
            for (int ks = 0; ks < 4; ++ks) {
                const int kb0 = ks * 32;
                const int kb1 = (ks + 4) * 32;
                // A fragment (16x32 f16): halves 0-7 = K kb+hi*8+0..7,
                //                         halves 8-15 = K kb+16+hi*8+0..7
                v8h a0lo = *(const v8h*)(arow + kb0 + hi * 8);
                v8h a0hi = *(const v8h*)(arow + kb0 + 16 + hi * 8);
                v16h a0 = __builtin_shufflevector(a0lo, a0hi,
                    0,1,2,3,4,5,6,7,8,9,10,11,12,13,14,15);
                v8h a1lo = *(const v8h*)(arow + kb1 + hi * 8);
                v8h a1hi = *(const v8h*)(arow + kb1 + 16 + hi * 8);
                v16h a1 = __builtin_shufflevector(a1lo, a1hi,
                    0,1,2,3,4,5,6,7,8,9,10,11,12,13,14,15);
                // B fragment (32x16): halves 0-7 = K kb+hi*8+0..7, 8-15 = K kb+16+hi*8+0..7 at column n0+m
                v16h b0 = __builtin_shufflevector(*(const v8h*)(bcol + kb0 + hi * 8),
                    *(const v8h*)(bcol + kb0 + 16 + hi * 8),
                    0,1,2,3,4,5,6,7,8,9,10,11,12,13,14,15);
                v16h b1 = __builtin_shufflevector(*(const v8h*)(bcol + kb1 + hi * 8),
                    *(const v8h*)(bcol + kb1 + 16 + hi * 8),
                    0,1,2,3,4,5,6,7,8,9,10,11,12,13,14,15);
                acc0 = __builtin_amdgcn_wmma_f32_16x16x32_f16(
                    false, a0, false, b0, (short)0, acc0, false, false);
                acc1 = __builtin_amdgcn_wmma_f32_16x16x32_f16(
                    false, a1, false, b1, (short)0, acc1, false, false);
            }
            const float bv = use_bias ? bias[n0 + m] : 0.0f; // bias only on value channel
            #pragma unroll
            for (int r = 0; r < 8; ++r) {
                const int mm = r + hi * 8;                   // C/D row layout
                sZ[(ch * TILE_P + mm) * HD + n0 + m] = acc0[r] + acc1[r] + bv;
            }
        }
    };

    // ---- coupled tanh forward-mode rule (reads cross-channel Z) ----
    auto activate = [&](int ch, int pLo, int pHi) {
        for (int p = pLo; p < pHi; ++p) {
            for (int j = lane; j < HD; j += 32) {
                const float zv = sZ[(0 * TILE_P + p) * HD + j];
                const float t  = tanhf(zv);
                const float s  = 1.0f - t * t;
                float a;
                if (ch == 0)      a = t;                                    // value
                else if (ch <= 3) a = s * sZ[(ch * TILE_P + p) * HD + j];   // tangents
                else {                                                      // pure 2nd
                    const int d = ch - 3;                                   // 1=x, 2=y
                    const float zd = sZ[(d * TILE_P + p) * HD + j];
                    a = s * sZ[(ch * TILE_P + p) * HD + j] - 2.0f * t * s * zd * zd;
                }
                sA[(ch * TILE_P + p) * HD + j] = (_Float16)a;
            }
        }
    };

    // ================= u-net (6 channels, one per wave) =================
    // layer 1 (3 -> 256): direct VALU, seed channels
    for (int p = 0; p < TILE_P; ++p) {
        const float x0 = X[(g0p + p) * 3 + 0];
        const float x1 = X[(g0p + p) * 3 + 1];
        const float x2 = X[(g0p + p) * 3 + 2];
        for (int j = lane; j < HD; j += 32) {
            const float w0 = Wu1[j], w1 = Wu1[HD + j], w2 = Wu1[2 * HD + j];
            const float z  = x0 * w0 + x1 * w1 + x2 * w2 + bu1[j];
            const float t  = tanhf(z), s = 1.0f - t * t;
            float a;
            switch (wv) {
                case 0:  a = t;                       break; // value
                case 1:  a = s * w0;                  break; // d/dx
                case 2:  a = s * w1;                  break; // d/dy
                case 3:  a = s * w2;                  break; // d/dt
                case 4:  a = -2.0f * t * s * w0 * w0; break; // d2/dx2
                default: a = -2.0f * t * s * w1 * w1; break; // d2/dy2
            }
            sA[(wv * TILE_P + p) * HD + j] = (_Float16)a;
        }
    }
    __syncthreads();

    // layer 2
    if (wv == 0) tdm_stage_W(WT + 0 * HD * HD, sW);
    __syncthreads();
    gemm(wv, 0, 16, bu2, wv == 0);
    __syncthreads();
    activate(wv, 0, TILE_P);
    __syncthreads();

    // layer 3
    if (wv == 0) tdm_stage_W(WT + 1 * HD * HD, sW);
    __syncthreads();
    gemm(wv, 0, 16, bu3, wv == 0);
    __syncthreads();
    activate(wv, 0, TILE_P);
    __syncthreads();

    // layer 4 (256 -> 2): one dot per (channel, point, output)
    {
        const int p = lane >> 1, jo = lane & 1;
        float acc = (wv == 0) ? bu4[jo] : 0.0f;
        const _Float16* arow = &sA[(wv * TILE_P + p) * HD];
        for (int k = 0; k < HD; ++k) acc += (float)arow[k] * Wu4[k * 2 + jo];
        int slot;
        if (wv == 0)      slot = jo;                         // u, v
        else if (wv <= 3) slot = (jo ? 5 : 2) + (wv - 1);    // Ju*, Jv*
        else if (wv == 4) slot = 8 + jo;                     // Hux, Hvx
        else              slot = 10 + jo;                    // Huy, Hvy
        sres[p][slot] = acc;
    }
    __syncthreads();

    // ================= p-net (3 channels, split across all 6 waves) =================
    // wave w -> channel w%3; waves 0-2 take half 0, waves 3-5 take half 1
    const int pch  = (wv < 3) ? wv : wv - 3;
    const int half = (wv < 3) ? 0 : 1;

    // layer 1: split points across wave halves
    for (int p = half * 8; p < half * 8 + 8; ++p) {
        const float x0 = X[(g0p + p) * 3 + 0];
        const float x1 = X[(g0p + p) * 3 + 1];
        const float x2 = X[(g0p + p) * 3 + 2];
        for (int j = lane; j < HD; j += 32) {
            const float w0 = Wp1[j], w1 = Wp1[HD + j], w2 = Wp1[2 * HD + j];
            const float z  = x0 * w0 + x1 * w1 + x2 * w2 + bp1[j];
            const float t  = tanhf(z), s = 1.0f - t * t;
            const float a  = (pch == 0) ? t : (pch == 1 ? s * w0 : s * w1);
            sA[(pch * TILE_P + p) * HD + j] = (_Float16)a;
        }
    }
    __syncthreads();

    if (wv == 0) tdm_stage_W(WT + 2 * HD * HD, sW);
    __syncthreads();
    gemm(pch, half * 8, half * 8 + 8, bp2, pch == 0);       // split N-tiles
    __syncthreads();
    activate(pch, half * 8, half * 8 + 8);                  // split points
    __syncthreads();

    if (wv == 0) tdm_stage_W(WT + 3 * HD * HD, sW);
    __syncthreads();
    gemm(pch, half * 8, half * 8 + 8, bp3, pch == 0);
    __syncthreads();
    activate(pch, half * 8, half * 8 + 8);
    __syncthreads();

    // p-net layer 4 (256 -> 1): need dp/dx (ch1) and dp/dy (ch2)
    if (wv == 0) {
        const int p = m, ch = 1 + hi;          // lanes 0-15: dp/dx, 16-31: dp/dy
        const _Float16* arow = &sA[(ch * TILE_P + p) * HD];
        float acc = 0.0f;
        for (int k = 0; k < HD; ++k) acc += (float)arow[k] * Wp4[k];
        sres[p][12 + hi] = acc;
    }
    __syncthreads();

    // ================= residual combine =================
    if (tid < TILE_P) {
        const int p = tid;
        const float u   = sres[p][0],  v   = sres[p][1];
        const float Jux = sres[p][2],  Juy = sres[p][3],  Jut = sres[p][4];
        const float Jvx = sres[p][5],  Jvy = sres[p][6],  Jvt = sres[p][7];
        const float Hux = sres[p][8],  Hvx = sres[p][9];
        const float Huy = sres[p][10], Hvy = sres[p][11];
        const float px  = sres[p][12], py  = sres[p][13];
        const float inv_re = 0.01f;
        out[g0p + p]            = Jut + u * Jux + v * Juy + px - inv_re * (Hux + Huy);
        out[NPTS + g0p + p]     = Jvt + u * Jvx + v * Jvy + py - inv_re * (Hvx + Hvy);
        out[2 * NPTS + g0p + p] = Jux + Jvy;
    }
}

extern "C" void kernel_launch(void* const* d_in, const int* in_sizes, int n_in,
                              void* d_out, int out_size, void* d_ws, size_t ws_size,
                              hipStream_t stream) {
    const float* X   = (const float*)d_in[0];
    const float* Wu1 = (const float*)d_in[1];  const float* bu1 = (const float*)d_in[2];
    const float* Wu2 = (const float*)d_in[3];  const float* bu2 = (const float*)d_in[4];
    const float* Wu3 = (const float*)d_in[5];  const float* bu3 = (const float*)d_in[6];
    const float* Wu4 = (const float*)d_in[7];  const float* bu4 = (const float*)d_in[8];
    const float* Wp1 = (const float*)d_in[9];  const float* bp1 = (const float*)d_in[10];
    const float* Wp2 = (const float*)d_in[11]; const float* bp2 = (const float*)d_in[12];
    const float* Wp3 = (const float*)d_in[13]; const float* bp3 = (const float*)d_in[14];
    const float* Wp4 = (const float*)d_in[15];
    (void)in_sizes; (void)n_in; (void)out_size;

    // d_ws: 4 x 256x256 f16 transposed weights = 512 KB
    _Float16* WT = (_Float16*)d_ws; (void)ws_size;

    cvt_weights<<<(4 * HD * HD + 255) / 256, 256, 0, stream>>>(Wu2, Wu3, Wp2, Wp3, WT);
    pinn_residual_kernel<<<NPTS / TILE_P, THREADS, 0, stream>>>(
        X, Wu1, bu1, bu2, bu3, Wu4, bu4, Wp1, bp1, bp2, bp3, Wp4, WT, (float*)d_out);
}
